// CIFM_63780264345953
// MI455X (gfx1250) — compile-verified, matrix operand order checked by reading, not run on hardware
//
#include <hip/hip_runtime.h>
#include <hip/hip_bf16.h>

typedef __bf16 bf16;
typedef __attribute__((ext_vector_type(16))) __bf16 v16bf;
typedef __attribute__((ext_vector_type(8)))  float  v8f;
typedef int i32x4 __attribute__((vector_size(16)));   // matches builtin's 'int __vector(4)' param

#define BATCH 8
#define SEQ   2048
#define DIM   512
#define ODIM  512
#define NEG_INF (-3.402823466e38f)

#ifndef __has_builtin
#define __has_builtin(x) 0
#endif

#if __has_builtin(__builtin_amdgcn_global_load_async_to_lds_b128)
#define HAVE_ASYNC_LDS 1
#else
#define HAVE_ASYNC_LDS 0
#endif

union V16U { v16bf v; uint4 q[2]; };

__device__ __forceinline__ v8f wmma_bf16(v16bf a, v16bf b, v8f c) {
  // D = A(16x32 bf16) * B(32x16 bf16) + C(16x16 f32)
  return __builtin_amdgcn_wmma_f32_16x16x32_bf16(false, a, false, b, (short)0, c, false, false);
}

__device__ __forceinline__ v8f zero8() {
  v8f z = {0.f, 0.f, 0.f, 0.f, 0.f, 0.f, 0.f, 0.f};
  return z;
}

// Per-lane bf16 operand fetch per CDNA5 16-bit A/B layout:
// caller passes p = row_base + kchunk_base + h*8 ; elements [0..7] at p, [8..15] at p+16.
__device__ __forceinline__ v16bf load_op(const bf16* p) {
  V16U u;
  u.q[0] = *(const uint4*)(p);
  u.q[1] = *(const uint4*)(p + 16);
  return u.v;
}

// 16-byte global -> LDS copy; async (ASYNCcnt-tracked, no VGPR round trip) when available.
__device__ __forceinline__ void copy_b128_to_lds(const bf16* gsrc, bf16* ldst) {
#if HAVE_ASYNC_LDS
  __builtin_amdgcn_global_load_async_to_lds_b128((i32x4*)(const_cast<bf16*>(gsrc)),
                                                 (i32x4*)(ldst), 0, 0);
#else
  *(uint4*)ldst = *(const uint4*)gsrc;
#endif
}

__device__ __forceinline__ void async_copies_done() {
#if HAVE_ASYNC_LDS
#if __has_builtin(__builtin_amdgcn_s_wait_asynccnt)
  __builtin_amdgcn_s_wait_asynccnt(0);
#else
  asm volatile("s_wait_asynccnt 0x0" ::: "memory");
#endif
#endif
}

// ---------------- pre-pass: fp32 -> bf16 conversions ----------------

__global__ void cvt_kernel(const float* __restrict__ src, bf16* __restrict__ dst, int n) {
  for (int i = blockIdx.x * blockDim.x + threadIdx.x; i < n; i += gridDim.x * blockDim.x)
    dst[i] = (bf16)src[i];
}

// VT[b][d][n] = V[b][n][d]  (bf16)
__global__ void cvt_vt_kernel(const float* __restrict__ V, bf16* __restrict__ VT) {
  const int total = BATCH * DIM * SEQ;
  for (int i = blockIdx.x * blockDim.x + threadIdx.x; i < total; i += gridDim.x * blockDim.x) {
    int n = i % SEQ;
    int d = (i / SEQ) % DIM;
    int b = i / (SEQ * DIM);
    VT[i] = (bf16)V[((size_t)b * SEQ + n) * DIM + d];
  }
}

// ---------------- pass 1: softmax row stats (exact two-pass) ----------------

__global__ void __launch_bounds__(128)
stats_kernel(const float* __restrict__ Q, const bf16* __restrict__ Kb,
             const float* __restrict__ scale_param,
             float* __restrict__ wsm, float* __restrict__ wsl) {
  __shared__ __align__(16) bf16 sQ[16 * DIM];
  __shared__ float sM[4][16];
  __shared__ float sL[4][16];

  const int b    = blockIdx.x >> 7;         // 128 row-tiles per batch
  const int row0 = (blockIdx.x & 127) << 4;
  const int tid  = threadIdx.x;
  const int wave = tid >> 5;
  const int lane = tid & 31;
  const int h    = lane >> 4;
  const int r    = lane & 15;

  const float c = scale_param[0] * 0.04419417382415922f; // 1/sqrt(512)

  const float* Qrow = Q + ((size_t)b * SEQ + row0) * DIM;
  for (int e = tid; e < 16 * DIM; e += 128) sQ[e] = (bf16)Qrow[e];
  __syncthreads();

  v16bf qa[16];
#pragma unroll
  for (int kc = 0; kc < 16; ++kc) qa[kc] = load_op(&sQ[r * DIM + kc * 32 + h * 8]);

  float m[8], l[8];
#pragma unroll
  for (int i = 0; i < 8; ++i) { m[i] = NEG_INF; l[i] = 0.f; }

  const bf16* Kbase = Kb + (size_t)b * SEQ * DIM;
  for (int t = wave; t < SEQ / 16; t += 4) {   // disjoint key tiles per wave
    const int j0 = t * 16;
    v8f acc = zero8();
#pragma unroll
    for (int kc = 0; kc < 16; ++kc) {
      v16bf Bm = load_op(Kbase + (size_t)(j0 + r) * DIM + kc * 32 + h * 8);
      acc = wmma_bf16(qa[kc], Bm, acc);
    }
    // C-layout: element (i,lane) = row i+8h, col = lane&15. Reduce over 16-lane col groups.
    float sc[8], tm[8];
#pragma unroll
    for (int i = 0; i < 8; ++i) { sc[i] = acc[i] * c; tm[i] = sc[i]; }
#pragma unroll
    for (int dm = 1; dm < 16; dm <<= 1) {
#pragma unroll
      for (int i = 0; i < 8; ++i) tm[i] = fmaxf(tm[i], __shfl_xor(tm[i], dm, 32));
    }
    float ts[8];
#pragma unroll
    for (int i = 0; i < 8; ++i) {
      float nm = fmaxf(m[i], tm[i]);
      ts[i] = __expf(sc[i] - nm);
      l[i] *= __expf(m[i] - nm);
      m[i]  = nm;
    }
#pragma unroll
    for (int dm = 1; dm < 16; dm <<= 1) {
#pragma unroll
      for (int i = 0; i < 8; ++i) ts[i] += __shfl_xor(ts[i], dm, 32);
    }
#pragma unroll
    for (int i = 0; i < 8; ++i) l[i] += ts[i];
  }

  if (r == 0) {  // lanes 0 and 16 hold full stats for rows 0..7 / 8..15
#pragma unroll
    for (int i = 0; i < 8; ++i) { sM[wave][i + 8 * h] = m[i]; sL[wave][i + 8 * h] = l[i]; }
  }
  __syncthreads();
  if (tid < 16) {
    float mm = NEG_INF;
#pragma unroll
    for (int w = 0; w < 4; ++w) mm = fmaxf(mm, sM[w][tid]);
    float ll = 0.f;
#pragma unroll
    for (int w = 0; w < 4; ++w) ll += sL[w][tid] * __expf(sM[w][tid] - mm);
    wsm[(size_t)b * SEQ + row0 + tid] = mm;
    wsl[(size_t)b * SEQ + row0 + tid] = ll;
  }
}

// ---------------- pass 2: attention + fused linear epilogue ----------------

__global__ void __launch_bounds__(128)
attn_kernel(const float* __restrict__ Q, const bf16* __restrict__ Kb,
            const float* __restrict__ V, const bf16* __restrict__ VT,
            const bf16* __restrict__ Wb,
            const float* __restrict__ wsm, const float* __restrict__ wsl,
            const float* __restrict__ scale_param,
            const float* __restrict__ ap, const float* __restrict__ bp,
            float* __restrict__ out) {
  extern __shared__ __align__(16) char smem[];
  bf16* sQ  = (bf16*)smem;            // 16 x 512  (16 KB)
  bf16* sK0 = sQ + 16 * DIM;          // 64 x 512  (64 KB)  double-buffered
  bf16* sK1 = sK0 + 64 * DIM;         // 64 x 512  (64 KB)
  bf16* sP  = sK1 + 64 * DIM;         // 16 x 64   ( 2 KB)
  bf16* sT  = sQ;                     // alias: sQ is dead after operand hoist

  const int b    = blockIdx.x >> 7;
  const int row0 = (blockIdx.x & 127) << 4;
  const int tid  = threadIdx.x;
  const int wave = tid >> 5;
  const int lane = tid & 31;
  const int h    = lane >> 4;
  const int r    = lane & 15;

  const float c  = scale_param[0] * 0.04419417382415922f;
  const float av = ap[0];
  const float bv = bp[0];

  const bf16* Kbase  = Kb + (size_t)b * SEQ * DIM;
  const bf16* VTbase = VT + (size_t)b * DIM * SEQ;

  // kick off async staging of K block 0 while we convert/stage Q
  for (int e = tid; e < 64 * DIM / 8; e += 128)
    copy_b128_to_lds(Kbase + (size_t)e * 8, sK0 + (size_t)e * 8);

  const float* Qrow = Q + ((size_t)b * SEQ + row0) * DIM;
  for (int e = tid; e < 16 * DIM; e += 128) sQ[e] = (bf16)Qrow[e];

  float mrow[8], linv[8];
#pragma unroll
  for (int i = 0; i < 8; ++i) {
    const int row = i + 8 * h;
    mrow[i] = wsm[(size_t)b * SEQ + row0 + row];
    linv[i] = 1.f / wsl[(size_t)b * SEQ + row0 + row];
  }
  __syncthreads();

  // hoist Q A-operands into registers (16 chunks x 8 VGPRs)
  v16bf qa[16];
#pragma unroll
  for (int kc = 0; kc < 16; ++kc) qa[kc] = load_op(&sQ[r * DIM + kc * 32 + h * 8]);

  v8f O[8];
#pragma unroll
  for (int t = 0; t < 8; ++t) O[t] = zero8();

  for (int jb = 0; jb < SEQ / 64; ++jb) {
    const int j0 = jb * 64;
    bf16* cur = (jb & 1) ? sK1 : sK0;
    bf16* nxt = (jb & 1) ? sK0 : sK1;

    async_copies_done();   // my async copies into `cur` have landed
    __syncthreads();       // everyone's copies have landed; prior readers of `nxt` done

    if (jb + 1 < SEQ / 64) {  // overlap next K block copy with this block's WMMAs
      const bf16* src = Kbase + (size_t)(j0 + 64) * DIM;
      for (int e = tid; e < 64 * DIM / 8; e += 128)
        copy_b128_to_lds(src + (size_t)e * 8, nxt + (size_t)e * 8);
    }

    // S tile: wave handles score columns j0 + wave*16 .. +16
    v8f S = zero8();
#pragma unroll
    for (int kc = 0; kc < 16; ++kc) {
      v16bf Bm = load_op(&cur[(wave * 16 + r) * DIM + kc * 32 + h * 8]);
      S = wmma_bf16(qa[kc], Bm, S);
    }
    // P = exp(S*c - m)/l  -> LDS (layout transpose C -> A operand)
#pragma unroll
    for (int i = 0; i < 8; ++i) {
      float p = __expf(S[i] * c - mrow[i]) * linv[i];
      sP[(i + 8 * h) * 64 + wave * 16 + r] = (bf16)p;
    }
    __syncthreads();

    // O[:, wave's 128-wide D slice] += P(16x64) @ V(64 x Dslice)
#pragma unroll
    for (int t = 0; t < 8; ++t) {
      const int dbase = wave * 128 + t * 16;
#pragma unroll
      for (int kc = 0; kc < 2; ++kc) {
        v16bf A  = load_op(&sP[r * 64 + kc * 32 + h * 8]);
        v16bf Bm = load_op(VTbase + (size_t)(dbase + r) * SEQ + j0 + kc * 32 + h * 8);
        O[t] = wmma_bf16(A, Bm, O[t]);
      }
    }
  }

  // epilogue: R = relu(O); T = V - R (bf16 into sT)
  float R[8][8];
#pragma unroll
  for (int t = 0; t < 8; ++t) {
    const int dcol = wave * 128 + t * 16 + r;
#pragma unroll
    for (int i = 0; i < 8; ++i) {
      const int row = i + 8 * h;
      float rv = fmaxf(O[t][i], 0.f);
      R[t][i] = rv;
      float vv = V[((size_t)b * SEQ + row0 + row) * DIM + dcol];
      sT[row * DIM + dcol] = (bf16)(vv - rv);
    }
  }
  __syncthreads();

  // C = relu(T @ W^T); out = a*R + b*C  (wave owns output cols wave*128 .. +128)
#pragma unroll
  for (int t = 0; t < 8; ++t) {
    const int obase = wave * 128 + t * 16;
    v8f Cacc = zero8();
#pragma unroll
    for (int kc = 0; kc < 16; ++kc) {
      const int kb = kc * 32;
      v16bf A  = load_op(&sT[r * DIM + kb + h * 8]);
      v16bf Bm = load_op(Wb + (size_t)(obase + r) * DIM + kb + h * 8);
      Cacc = wmma_bf16(A, Bm, Cacc);
    }
#pragma unroll
    for (int i = 0; i < 8; ++i) {
      const int row  = i + 8 * h;
      const int ocol = obase + r;
      float cv = fmaxf(Cacc[i], 0.f);
      out[((size_t)b * SEQ + row0 + row) * ODIM + ocol] = av * R[t][i] + bv * cv;
    }
  }
}

// ---------------- host ----------------

extern "C" void kernel_launch(void* const* d_in, const int* in_sizes, int n_in,
                              void* d_out, int out_size, void* d_ws, size_t ws_size,
                              hipStream_t stream) {
  (void)in_sizes; (void)n_in; (void)out_size; (void)ws_size;
  const float* Q  = (const float*)d_in[0];
  const float* K  = (const float*)d_in[1];
  const float* V  = (const float*)d_in[2];
  const float* W  = (const float*)d_in[3];
  const float* sc = (const float*)d_in[4];
  const float* a  = (const float*)d_in[5];
  const float* b  = (const float*)d_in[6];
  float* out = (float*)d_out;

  char* ws = (char*)d_ws;
  const size_t kv_bytes = (size_t)BATCH * SEQ * DIM * sizeof(bf16); // 16 MB each
  bf16*  Kb  = (bf16*)ws;                                  // [B][N][D]
  bf16*  VT  = (bf16*)(ws + kv_bytes);                     // [B][D][N]
  bf16*  Wb  = (bf16*)(ws + 2 * kv_bytes);                 // [O][D]
  float* wsm = (float*)(ws + 2 * kv_bytes + (size_t)ODIM * DIM * sizeof(bf16));
  float* wsl = wsm + (size_t)BATCH * SEQ;

  cvt_kernel<<<4096, 256, 0, stream>>>(K, Kb, BATCH * SEQ * DIM);
  cvt_vt_kernel<<<4096, 256, 0, stream>>>(V, VT);
  cvt_kernel<<<512, 256, 0, stream>>>(W, Wb, ODIM * DIM);

  stats_kernel<<<BATCH * (SEQ / 16), 128, 0, stream>>>(Q, Kb, sc, wsm, wsl);

  const int smem_bytes = (16 * DIM + 2 * 64 * DIM + 16 * 64) * (int)sizeof(bf16); // 149504
  (void)hipFuncSetAttribute(reinterpret_cast<const void*>(&attn_kernel),
                            hipFuncAttributeMaxDynamicSharedMemorySize, smem_bytes);
  attn_kernel<<<BATCH * (SEQ / 16), 128, smem_bytes, stream>>>(
      Q, Kb, V, VT, Wb, wsm, wsl, sc, a, b, out);
}
